// AttentionMechanism_1194000908409
// MI455X (gfx1250) — compile-verified, matrix operand order checked by reading, not run on hardware
//
#include <hip/hip_runtime.h>
#include <math.h>

typedef __attribute__((ext_vector_type(16))) _Float16 v16h;
typedef __attribute__((ext_vector_type(8)))  float    v8f;

#define B_SZ 32
#define L_SZ 4096
#define H_SZ 512
#define V_SZ 512
#define E_SZ 512   // BETA

#define TILE_M 128
#define LDS_ROW_U32 260   // 256 f16-pairs per row + 4 u32 pad (bank-conflict stagger)

// ---------------- fast transcendentals (branchless, no EXEC divergence) ----------------

__device__ __forceinline__ float fast_exp(float x) {
#if __has_builtin(__builtin_amdgcn_exp2f)
  return __builtin_amdgcn_exp2f(x * 1.4426950408889634f);   // v_exp_f32
#else
  return __expf(x);
#endif
}

__device__ __forceinline__ float fast_tanh(float x) {
#if __has_builtin(__builtin_amdgcn_tanhf)
  return __builtin_amdgcn_tanhf(x);                         // v_tanh_f32 (gfx1250)
#elif __has_builtin(__builtin_amdgcn_exp2f) && __has_builtin(__builtin_amdgcn_rcpf)
  // tanh(x) = 1 - 2/(1 + 2^(2*log2(e)*x)) ; exact limits at +/-inf, branch-free
  float t = __builtin_amdgcn_exp2f(x * 2.8853900817779268f);
  return 1.0f - 2.0f * __builtin_amdgcn_rcpf(1.0f + t);
#else
  float t = __expf(2.0f * x);
  return 1.0f - 2.0f / (1.0f + t);
#endif
}

// ---------------- small prep kernels ----------------

__global__ void k_convert_w(const float* __restrict__ W_w, _Float16* __restrict__ W16) {
  int idx = blockIdx.x * blockDim.x + threadIdx.x;   // 262144 total
  W16[idx] = (_Float16)W_w[idx];
}

__global__ void k_proj_h(const float* __restrict__ h_t, const float* __restrict__ W_h_w,
                         float* __restrict__ proj) {
  int idx = blockIdx.x * blockDim.x + threadIdx.x;   // 0..16383
  int b = idx >> 9;        // / 512
  int e = idx & 511;
  const float* hr = h_t  + b * H_SZ;
  const float* wr = W_h_w + e * H_SZ;
  float acc = 0.f;
#pragma unroll 4
  for (int h = 0; h < H_SZ; ++h) acc += hr[h] * wr[h];
  proj[idx] = acc;
}

// ---------------- main fused energy kernel (WMMA) ----------------
// E[b,l] = sum_e beta[e] * tanh(proj_h[b,e] + sum_v V_new[b,l,v] * W_w[e,v])

__global__ void k_energy(const float* __restrict__ Vn, const _Float16* __restrict__ W16,
                         const float* __restrict__ proj, const float* __restrict__ beta,
                         float* __restrict__ E) {
  extern __shared__ unsigned lds[];          // TILE_M * LDS_ROW_U32 u32 (f16 pairs)
  const int b   = blockIdx.y;
  const int l0  = blockIdx.x * TILE_M;
  const int tid = threadIdx.x;

  // Stage V_new[b, l0..l0+127, 0..511] -> f16 in LDS (coalesced f32x2 loads)
  {
    const float2* src = (const float2*)(Vn + ((size_t)(b * L_SZ + l0)) * V_SZ);
#pragma unroll 4
    for (int i = 0; i < 128; ++i) {          // 128 iters * 256 threads = 32768 pairs
      int p   = i * 256 + tid;
      int row = p >> 8;
      int kp  = p & 255;
      float2 f = src[row * 256 + kp];
      union { _Float16 h[2]; unsigned u; } cv;
      cv.h[0] = (_Float16)f.x; cv.h[1] = (_Float16)f.y;
      lds[row * LDS_ROW_U32 + kp] = cv.u;
    }
  }
  __syncthreads();

  const int lane = tid & 31;
  const int wave = tid >> 5;
  const int m0   = wave * 16;               // 8 waves x 16 rows = 128 rows
  const int g    = lane >> 4;               // lane half (K-group select)
  const int n    = lane & 15;               // A: row-in-tile, B: col-in-tile
  const unsigned* Wu = (const unsigned*)W16;

  // K-pair-within-32 offset (in u32 units) for fragment VGPR i, half g
  int kpos[8];
#pragma unroll
  for (int i = 0; i < 8; ++i)
    kpos[i] = (i < 4) ? (i + 4 * g) : (8 + (i - 4) + 4 * g);

  const unsigned* aBase = lds + (m0 + n) * LDS_ROW_U32;

  float rsum[8];
#pragma unroll
  for (int r = 0; r < 8; ++r) rsum[r] = 0.f;

  for (int nt = 0; nt < E_SZ / 16; ++nt) {
    const int e = nt * 16 + n;              // BETA column owned by this lane
    v8f acc = {};
#pragma unroll 2
    for (int ks = 0; ks < V_SZ / 32; ++ks) {
      union { unsigned u[8]; v16h v; } af, bf;
#pragma unroll
      for (int i = 0; i < 8; ++i) {
        int kp = ks * 16 + kpos[i];
        af.u[i] = aBase[kp];                // A: V_new tile from LDS
        bf.u[i] = Wu[e * 256 + kp];         // B: W_w row (L2-resident)
      }
      acc = __builtin_amdgcn_wmma_f32_16x16x32_f16(
          false, af.v, false, bf.v, (short)0, acc, false, false);
    }
    // fused epilogue: hw tanh + beta weighting, accumulate partial E per row
    const float ph = proj[b * E_SZ + e];
    const float bt = beta[e];
#pragma unroll
    for (int r = 0; r < 8; ++r)
      rsum[r] += bt * fast_tanh(acc[r] + ph);
  }

  // Reduce the 16 lanes of each half (lanes 0-15 -> row r, 16-31 -> row r+8)
#pragma unroll
  for (int r = 0; r < 8; ++r) {
    float v = rsum[r];
    v += __shfl_xor(v, 1);
    v += __shfl_xor(v, 2);
    v += __shfl_xor(v, 4);
    v += __shfl_xor(v, 8);
    rsum[r] = v;
  }
  if ((lane & 15) == 0) {
    int rowbase = l0 + m0 + g * 8;
#pragma unroll
    for (int r = 0; r < 8; ++r)
      E[b * L_SZ + rowbase + r] = rsum[r];
  }
}

// ---------------- softmax over L per batch ----------------

__global__ void k_softmax(const float* __restrict__ E, float* __restrict__ A) {
  __shared__ float red[256];
  const int b = blockIdx.x, tid = threadIdx.x;
  const float* Er = E + b * L_SZ;

  float m = -3.402823466e38f;
  for (int i = tid; i < L_SZ; i += 256) m = fmaxf(m, Er[i]);
  red[tid] = m; __syncthreads();
  for (int s = 128; s > 0; s >>= 1) {
    if (tid < s) red[tid] = fmaxf(red[tid], red[tid + s]);
    __syncthreads();
  }
  m = red[0]; __syncthreads();

  float sum = 0.f;
  for (int i = tid; i < L_SZ; i += 256) sum += fast_exp(Er[i] - m);
  red[tid] = sum; __syncthreads();
  for (int s = 128; s > 0; s >>= 1) {
    if (tid < s) red[tid] += red[tid + s];
    __syncthreads();
  }
  const float inv = 1.0f / red[0];

  for (int i = tid; i < L_SZ; i += 256)
    A[b * L_SZ + i] = fast_exp(Er[i] - m) * inv;
}

// ---------------- context: C_t[v,b] = sum_l A[b,l] * V_new[b,l,v] ----------------

__global__ void k_context(const float* __restrict__ Vn, const float* __restrict__ A,
                          float* __restrict__ C) {
  const int b = blockIdx.x;
  const int v = threadIdx.x;                // 512 threads = full V row, coalesced
  const float* Ar = A + b * L_SZ;
  const float* Vr = Vn + (size_t)b * L_SZ * V_SZ + v;
  float acc = 0.f;
  for (int l = 0; l < L_SZ; ++l)
    acc += Ar[l] * Vr[(size_t)l * V_SZ];
  C[v * B_SZ + b] = acc;                    // transposed [V, B]
}

// ---------------- launcher ----------------

extern "C" void kernel_launch(void* const* d_in, const int* in_sizes, int n_in,
                              void* d_out, int out_size, void* d_ws, size_t ws_size,
                              hipStream_t stream) {
  const float* Vn    = (const float*)d_in[0];
  const float* h_t   = (const float*)d_in[1];
  const float* W_h_w = (const float*)d_in[2];
  const float* W_w   = (const float*)d_in[3];
  const float* beta  = (const float*)d_in[4];

  char* ws = (char*)d_ws;
  float*    proj = (float*)ws;                          // 32*512*4   = 65536 B
  _Float16* W16  = (_Float16*)(ws + 65536);             // 512*512*2  = 524288 B
  float*    E    = (float*)(ws + 65536 + 524288);       // 32*4096*4  = 524288 B

  float* C = (float*)d_out;                 // [V, B]  = 16384 floats
  float* A = (float*)d_out + V_SZ * B_SZ;   // [B, 1, L] = 131072 floats

  k_convert_w<<<1024, 256, 0, stream>>>(W_w, W16);
  k_proj_h  <<<64,   256, 0, stream>>>(h_t, W_h_w, proj);

  dim3 g2(L_SZ / TILE_M, B_SZ);
  size_t smem = (size_t)TILE_M * LDS_ROW_U32 * 4;       // 133120 B (< 320 KB WGP LDS)
  k_energy  <<<g2, 256, smem, stream>>>(Vn, W16, proj, beta, E);

  k_softmax <<<B_SZ, 256, 0, stream>>>(E, A);
  k_context <<<B_SZ, V_SZ, 0, stream>>>(Vn, A, C);
}
